// Structure2Vec_54365696033293
// MI455X (gfx1250) — matrix operand order repaired and checked by reading, not compiled, not run on hardware
//
#include <hip/hip_runtime.h>

// Structure2Vec fused kernel for gfx1250 (MI455X, wave32, WMMA).
//
// out[b,j,p] = relu( x[b,j]*W1[p] + (adj^T @ mu @ W2)[j,p]
//                    + pos[b,j]*v_pos[p] + neg[b,j]*v_neg[p] + (b1+b2+b3)[p] )
// with v_pos = relu(t)@W3, v_neg = relu(-t)@W3 (unit4 is rank-2 -> no 3rd GEMM),
// pos/neg = column sums of relu(+/- weight*adj), fused into the adj stream.
//
// Main GEMM uses v_wmma_f32_16x16x32_f16 with hi/lo split of mu (adj is {0,1},
// exact in f16) -> fp32-grade accuracy while hitting the CDNA5 matrix pipe.

typedef _Float16 v16h __attribute__((ext_vector_type(16)));
typedef _Float16 v8h  __attribute__((ext_vector_type(8)));
typedef float    v8f  __attribute__((ext_vector_type(8)));

#define B_  32
#define N_  512
#define P_  128
#define BJ  64      // output columns (j) per block
#define KT  32      // WMMA K per step
#define LDA 40      // LDS row stride in halves (80B: 16B-aligned, conflict-breaking)
#define LDB 40

struct __align__(16) Smem {
  union U {
    struct T {
      _Float16 A  [BJ * LDA];   // adj^T tile  [j_local][i_local], f16 (exact)
      _Float16 Bhi[P_ * LDB];   // mu^T tile   [p][i_local], f16 high part
      _Float16 Blo[P_ * LDB];   // mu^T tile   [p][i_local], f16 low  part
    } t;
    float agg[BJ * P_];         // adj^T @ mu result (epilogue phase)
  } u;
  float posArr[4][BJ];
  float negArr[4][BJ];
  float pos[BJ];
  float neg[BJ];
};

// ---- prep: v_pos[p], v_neg[p], fused bias (b1+b2+b3) into workspace -------
__global__ __launch_bounds__(128) void s2v_prep(const float* __restrict__ W3,
                                                const float* __restrict__ theta4,
                                                const float* __restrict__ b1,
                                                const float* __restrict__ b2,
                                                const float* __restrict__ b3,
                                                float* __restrict__ ws) {
  const int p = threadIdx.x;
  float vp = 0.f, vn = 0.f;
  for (int q = 0; q < P_; ++q) {
    const float tq = theta4[q];
    const float w3 = W3[q * P_ + p];
    vp = fmaf(fmaxf(tq, 0.f), w3, vp);
    vn = fmaf(fmaxf(-tq, 0.f), w3, vn);
  }
  ws[p]          = vp;
  ws[P_ + p]     = vn;
  ws[2 * P_ + p] = b1[p] + b2[p] + b3[p];
}

// ---- main fused kernel: one block per (batch, 64-col tile) ----------------
__global__ __launch_bounds__(256) void s2v_main(
    const float* __restrict__ x,
    const float* __restrict__ mu,
    const float* __restrict__ weight,
    const float* __restrict__ adj,
    const float* __restrict__ W1,
    const float* __restrict__ W2,
    const float* __restrict__ ws,
    float* __restrict__ out) {
  __shared__ Smem sm;
  const int tid  = threadIdx.x;
  const int lane = tid & 31;
  const int wave = tid >> 5;          // 8 waves
  const int b     = blockIdx.y;
  const int jbase = blockIdx.x * BJ;

  // wave -> tile assignment: 4 M-tiles (j) x 2 N-halves; 4 N-tiles per wave
  const int jw   = wave & 3;          // M tile index (rows jw*16..+15)
  const int pw   = wave >> 2;         // N-half: tiles pw*4 .. pw*4+3
  const int nloc = lane & 15;
  const int h    = lane >> 4;         // lane half (K-group select per ISA layout)

  // staging maps
  const int js  = tid & 63;           // adj/weight: fixed local j per thread
  const int irs = tid >> 6;           // 0..3, 8 i-rows each
  const int ps  = tid & 127;          // mu: fixed p per thread
  const int ir2 = tid >> 7;           // 0..1, 16 i-rows each

  v8f c[4] = {};                      // f32 accumulators (4 N-tiles)
  float ppos = 0.f, pneg = 0.f;

  const long adjBase = (long)b * N_ * N_;
  const long muBase  = (long)b * N_ * P_;

  for (int kt = 0; kt < N_ / KT; ++kt) {
    const int i0 = kt * KT;

    // stage adj (->f16, exact) + fused pos/neg reduction over weight*adj
    #pragma unroll
    for (int k = 0; k < 8; ++k) {
      const int  i  = irs * 8 + k;
      const long gi = adjBase + (long)(i0 + i) * N_ + jbase + js;
      const float a = adj[gi];
      const float w = weight[gi] * a;
      ppos += fmaxf(w, 0.f);
      pneg += fmaxf(-w, 0.f);
      sm.u.t.A[js * LDA + i] = (_Float16)a;
    }
    // stage mu with hi/lo f16 split (transposed: K-contiguous)
    #pragma unroll
    for (int k = 0; k < 16; ++k) {
      const int   i  = ir2 * 16 + k;
      const float m  = mu[muBase + (long)(i0 + i) * P_ + ps];
      const _Float16 mh = (_Float16)m;
      const _Float16 ml = (_Float16)(m - (float)mh);
      sm.u.t.Bhi[ps * LDB + i] = mh;
      sm.u.t.Blo[ps * LDB + i] = ml;
    }
    __syncthreads();

    // A fragment (16x32 f16): lane<16 -> K 0..15; lane>=16 -> K 16..31
    const _Float16* ab = &sm.u.t.A[(jw * 16 + nloc) * LDA + h * 16];
    const v8h alo = *(const v8h*)ab;
    const v8h ahi = *(const v8h*)(ab + 8);
    v16h afrag;
    #pragma unroll
    for (int k = 0; k < 8; ++k) { afrag[k] = alo[k]; afrag[8 + k] = ahi[k]; }

    #pragma unroll
    for (int t = 0; t < 4; ++t) {
      const int pb = (pw * 4 + t) * 16;
      // B fragment (32x16 f16): lanes<16 hold K 0..15, lanes>=16 hold K 16..31
      const _Float16* bbh = &sm.u.t.Bhi[(pb + nloc) * LDB + h * 16];
      const v8h bh0 = *(const v8h*)bbh;
      const v8h bh1 = *(const v8h*)(bbh + 8);
      v16h bfh;
      #pragma unroll
      for (int k = 0; k < 8; ++k) { bfh[k] = bh0[k]; bfh[8 + k] = bh1[k]; }
      c[t] = __builtin_amdgcn_wmma_f32_16x16x32_f16(false, afrag, false, bfh,
                                                    (short)0, c[t], false, false);

      const _Float16* bbl = &sm.u.t.Blo[(pb + nloc) * LDB + h * 16];
      const v8h bl0 = *(const v8h*)bbl;
      const v8h bl1 = *(const v8h*)(bbl + 8);
      v16h bfl;
      #pragma unroll
      for (int k = 0; k < 8; ++k) { bfl[k] = bl0[k]; bfl[8 + k] = bl1[k]; }
      c[t] = __builtin_amdgcn_wmma_f32_16x16x32_f16(false, afrag, false, bfl,
                                                    (short)0, c[t], false, false);
    }
    __syncthreads();
  }

  // spill agg tiles to LDS (C/D layout: VGPR r -> row h*8+r, col nloc)
  #pragma unroll
  for (int t = 0; t < 4; ++t) {
    const int pb = (pw * 4 + t) * 16;
    #pragma unroll
    for (int r = 0; r < 8; ++r)
      sm.u.agg[(jw * 16 + h * 8 + r) * P_ + pb + nloc] = c[t][r];
  }
  sm.posArr[irs][js] = ppos;
  sm.negArr[irs][js] = pneg;
  __syncthreads();
  if (tid < BJ) {
    sm.pos[tid] = sm.posArr[0][tid] + sm.posArr[1][tid] + sm.posArr[2][tid] + sm.posArr[3][tid];
    sm.neg[tid] = sm.negArr[0][tid] + sm.negArr[1][tid] + sm.negArr[2][tid] + sm.negArr[3][tid];
  }
  __syncthreads();

  // epilogue: exact f32 (agg @ W2) + rank-2 unit3 + unit1 + biases, relu
  const int pe = tid & 127;
  const int jh = tid >> 7;            // thread owns 32 rows: j = jh + 2*jj
  const float w1p   = W1[pe];
  const float vpp   = ws[pe];
  const float vnp   = ws[P_ + pe];
  const float biasp = ws[2 * P_ + pe];

  float acc[32];
  #pragma unroll
  for (int jj = 0; jj < 32; ++jj) {
    const int j = jh + 2 * jj;
    const float xv = x[(long)b * N_ + jbase + j];
    acc[jj] = fmaf(xv, w1p, biasp) + sm.pos[j] * vpp + sm.neg[j] * vnp;
  }
  for (int q = 0; q < P_; ++q) {
    const float w2 = W2[q * P_ + pe];
    #pragma unroll
    for (int jj = 0; jj < 32; ++jj)
      acc[jj] = fmaf(sm.u.agg[(jh + 2 * jj) * P_ + q], w2, acc[jj]);
  }
  #pragma unroll
  for (int jj = 0; jj < 32; ++jj) {
    const int j = jh + 2 * jj;
    out[muBase + (long)(jbase + j) * P_ + pe] = fmaxf(acc[jj], 0.f);
  }
}

extern "C" void kernel_launch(void* const* d_in, const int* in_sizes, int n_in,
                              void* d_out, int out_size, void* d_ws, size_t ws_size,
                              hipStream_t stream) {
  (void)in_sizes; (void)n_in; (void)out_size; (void)ws_size;
  const float* x      = (const float*)d_in[0];
  const float* mu     = (const float*)d_in[1];
  const float* weight = (const float*)d_in[2];
  const float* adj    = (const float*)d_in[3];
  const float* W1     = (const float*)d_in[4];
  const float* b1     = (const float*)d_in[5];
  const float* W2     = (const float*)d_in[6];
  const float* b2     = (const float*)d_in[7];
  const float* W3     = (const float*)d_in[8];
  const float* b3     = (const float*)d_in[9];
  const float* theta4 = (const float*)d_in[10];
  float* ws  = (float*)d_ws;
  float* out = (float*)d_out;

  s2v_prep<<<1, 128, 0, stream>>>(W3, theta4, b1, b2, b3, ws);
  dim3 grid(N_ / BJ, B_);   // 8 x 32 = 256 blocks, 256 threads (8 wave32)
  s2v_main<<<grid, 256, 0, stream>>>(x, mu, weight, adj, W1, W2, ws, out);
}
